// Contrastive_loss_13434657702604
// MI455X (gfx1250) — compile-verified, matrix operand order checked by reading, not run on hardware
//
#include <hip/hip_runtime.h>

// ---------------------------------------------------------------------------
// Contrastive (NT-Xent style) loss for MI455X / gfx1250.
//
//   1) normalize rows, split fp32 -> bf16 hi + bf16 lo  (3-term bf16 WMMA
//      emulation of fp32 GEMM: hi*hi + hi*lo + lo*hi ~ fp32 accuracy)
//   2) three fused GEMM+exp+reduce passes over a.a^T, a.b^T, b.b^T using
//      v_wmma_f32_16x16x32_bf16; only row sums / col sums / diagonals of
//      exp(S/tau) survive (the NxN matrices are never materialized)
//   3) tiny finalize kernel: loss_i = 0.5*(log den1 + log den2) - log d_ab
//
// Double-buffered LDS staging driven by CDNA5 async global->LDS DMA
// (GLOBAL_LOAD_ASYNC_TO_LDS_B128 + s_wait_asynccnt), with a synchronous
// ds_store fallback if the builtins are unavailable.
// ---------------------------------------------------------------------------

typedef unsigned short u16;
typedef unsigned int   u32;

typedef __attribute__((ext_vector_type(16))) __bf16 v16bf;
typedef __attribute__((ext_vector_type(8)))  __bf16 v8bf;
typedef __attribute__((ext_vector_type(8)))  float  v8f;
typedef __attribute__((ext_vector_type(4)))  int    v4i;

#define N_ROWS   8192
#define D_DIM    512
#define KCHUNK   32
#define NKCHUNK  (D_DIM / KCHUNK)       // 16
#define LDS_PAD  8
#define LDS_STR  (KCHUNK + LDS_PAD)     // 40 u16 = 80 B row stride (16B aligned)
#define TILE_ELEMS (128 * LDS_STR)      // one 128x32 (padded) tile
#define BUF_ELEMS  (4 * TILE_ELEMS)     // Ahi,Alo,Bhi,Blo
#define INV_TAU  2.5f                   // 1 / 0.4

// ---- CDNA5 async global->LDS path (guarded; falls back to ds_store) -------
#if defined(__has_builtin)
#if __has_builtin(__builtin_amdgcn_global_load_async_to_lds_b128) && \
    __has_builtin(__builtin_amdgcn_s_wait_asynccnt)
#define USE_ASYNC_LDS 1
#endif
#endif

#ifdef USE_ASYNC_LDS
typedef __attribute__((address_space(1))) v4i* gv4i_p;
typedef __attribute__((address_space(3))) v4i* lv4i_p;
__device__ inline void cp16_async(const u16* g, u16* l) {
    __builtin_amdgcn_global_load_async_to_lds_b128(
        (gv4i_p)(u16*)g, (lv4i_p)l, 0, 0);
}
__device__ inline void wait_async_done() { __builtin_amdgcn_s_wait_asynccnt(0); }
#else
__device__ inline void cp16_async(const u16* g, u16* l) {
    *(uint4*)l = *(const uint4*)g;      // sync fallback (ds_store_b128)
}
__device__ inline void wait_async_done() {}
#endif

// ---- bf16 helpers (round-to-nearest-even) --------------------------------
__device__ inline u16 f2bf_rne(float f) {
    u32 u = __float_as_uint(f);
    u32 r = u + 0x7FFFu + ((u >> 16) & 1u);
    return (u16)(r >> 16);
}
__device__ inline float bf2f(u16 h) {
    return __uint_as_float(((u32)h) << 16);
}

// ---- WMMA wrapper ---------------------------------------------------------
__device__ inline v8f wmma_bf16(v16bf a, v16bf b, v8f c) {
    // (neg_a, A, neg_b, B, c_mod, C, reuse_a, reuse_b)
    return __builtin_amdgcn_wmma_f32_16x16x32_bf16(false, a, false, b,
                                                   (short)0, c, false, false);
}

// ---- fragment load from LDS (16-bit A layout, 16x32 tile, wave32) ---------
// lanes 0-15 row M=L, elems 0..7 -> K 0..7, elems 8..15 -> K 16..23
// lanes 16-31 row M=L-16, elems 0..7 -> K 8..15, elems 8..15 -> K 24..31
// B operand uses the same loader: it is Y's rows presented as columns
// (A-fragment-of-X == B-fragment-of-X^T per the ISA layout symmetry).
union FragU { v16bf v; v8bf h[2]; };

__device__ inline v16bf load_frag(const u16* tile, int row0, int lane) {
    const u16* p = tile + (row0 + (lane & 15)) * LDS_STR + ((lane >> 4) << 3);
    FragU f;
    f.h[0] = *reinterpret_cast<const v8bf*>(p);        // K base .. base+7
    f.h[1] = *reinterpret_cast<const v8bf*>(p + 16);   // K base+16 .. base+23
    return f.v;
}

// ---- stage one 128-row x 32-K chunk of all four operand parts -------------
__device__ inline void stage_chunk(const u16* __restrict__ Xhi,
                                   const u16* __restrict__ Xlo,
                                   const u16* __restrict__ Yhi,
                                   const u16* __restrict__ Ylo,
                                   u16* sbuf, int rowbase, int colbase,
                                   int kc, int tid) {
    u16* sAhi = sbuf;
    u16* sAlo = sbuf + 1 * TILE_ELEMS;
    u16* sBhi = sbuf + 2 * TILE_ELEMS;
    u16* sBlo = sbuf + 3 * TILE_ELEMS;
#pragma unroll
    for (int v = 0; v < 2; ++v) {
        int idx = tid + v * 256;          // 0..511 (512 x b128 per part)
        int r   = idx >> 2;               // 0..127
        int cg  = idx & 3;                // col group of 8 bf16
        size_t ga = (size_t)(rowbase + r) * D_DIM + kc + cg * 8;
        size_t gb = (size_t)(colbase + r) * D_DIM + kc + cg * 8;
        int    lo = r * LDS_STR + cg * 8;
        cp16_async(Xhi + ga, sAhi + lo);
        cp16_async(Xlo + ga, sAlo + lo);
        cp16_async(Yhi + gb, sBhi + lo);
        cp16_async(Ylo + gb, sBlo + lo);
    }
}

// ===========================================================================
// Kernel 1: per-row L2 normalize + split into bf16 hi/lo
// grid = (8192, 2), block = 256 ; blockIdx.y selects z1 or z2
// ===========================================================================
__global__ void __launch_bounds__(256)
normalize_split(const float* __restrict__ z1, const float* __restrict__ z2,
                u16* __restrict__ ahi, u16* __restrict__ alo,
                u16* __restrict__ bhi, u16* __restrict__ blo) {
    const float* z  = blockIdx.y ? z2  : z1;
    u16*         hi = blockIdx.y ? bhi : ahi;
    u16*         lo = blockIdx.y ? blo : alo;

    const int row = blockIdx.x;
    const int t   = threadIdx.x;
    const int lane = t & 31, wid = t >> 5;

    const float* zr = z + (size_t)row * D_DIM;
    float x0 = zr[t];
    float x1 = zr[t + 256];
    float ss = x0 * x0 + x1 * x1;
#pragma unroll
    for (int m = 16; m >= 1; m >>= 1) ss += __shfl_xor(ss, m, 32);

    __shared__ float red[8];
    __shared__ float s_scale;
    if (lane == 0) red[wid] = ss;
    __syncthreads();
    if (t == 0) {
        float tot = 0.f;
#pragma unroll
        for (int i = 0; i < 8; ++i) tot += red[i];
        float n = sqrtf(tot);
        s_scale = 1.0f / fmaxf(n, 1e-12f);   // matches F.normalize eps
    }
    __syncthreads();
    const float sc = s_scale;

    size_t base = (size_t)row * D_DIM;
#pragma unroll
    for (int e = 0; e < 2; ++e) {
        float a  = (e ? x1 : x0) * sc;
        u16   h  = f2bf_rne(a);
        float r  = a - bf2f(h);
        u16   l  = f2bf_rne(r);
        size_t idx = base + t + e * 256;
        hi[idx] = h;
        lo[idx] = l;
    }
}

// ===========================================================================
// Kernel 2: fused  exp(X*Y^T / tau)  ->  col sums (+ row sums) (+ diagonal)
// 128x128 output tile per block (256 thr = 8 waves, each wave 32x64).
// Split-bf16: acc += Ahi*Bhi + Ahi*Blo + Alo*Bhi   (3 WMMAs / K-step)
// Double-buffered async staging: chunk k+1 DMA overlaps chunk k WMMAs.
// ===========================================================================
__global__ void __launch_bounds__(256)
sim_gemm(const u16* __restrict__ Xhi, const u16* __restrict__ Xlo,
         const u16* __restrict__ Yhi, const u16* __restrict__ Ylo,
         float* __restrict__ colsum, float* __restrict__ rowsum,
         float* __restrict__ diag, int want_row) {
    const int tid  = threadIdx.x;
    const int lane = tid & 31;
    const int wid  = tid >> 5;
    const int rowbase = blockIdx.y * 128;
    const int colbase = blockIdx.x * 128;
    const int wr0 = (wid & 3) * 32;   // wave row offset inside block tile
    const int wc0 = (wid >> 2) * 64;  // wave col offset inside block tile

    __shared__ __align__(16) u16 lds[2 * BUF_ELEMS];   // 2 x 40 KB

    v8f acc[2][4];
#pragma unroll
    for (int mt = 0; mt < 2; ++mt)
#pragma unroll
        for (int nt = 0; nt < 4; ++nt) acc[mt][nt] = (v8f){};

    // prologue: kick off chunk 0 into buffer 0
    stage_chunk(Xhi, Xlo, Yhi, Ylo, lds, rowbase, colbase, 0, tid);

    for (int i = 0; i < NKCHUNK; ++i) {
        // my async copies into buf[i&1] done, then all waves' copies visible
        wait_async_done();
        __syncthreads();

        // overlap: start DMA of chunk i+1 into the other buffer.
        // Safe: every wave finished READING that buffer before the barrier
        // above (its reads happened in iteration i-1).
        if (i + 1 < NKCHUNK)
            stage_chunk(Xhi, Xlo, Yhi, Ylo, lds + ((i + 1) & 1) * BUF_ELEMS,
                        rowbase, colbase, (i + 1) * KCHUNK, tid);

        u16* sbuf = lds + (i & 1) * BUF_ELEMS;
        u16* sAhi = sbuf;
        u16* sAlo = sbuf + 1 * TILE_ELEMS;
        u16* sBhi = sbuf + 2 * TILE_ELEMS;
        u16* sBlo = sbuf + 3 * TILE_ELEMS;

        // --- 8 x (16x16) C tiles per wave, 3 WMMAs each = 24 per chunk ---
        v16bf ah[2], al[2];
#pragma unroll
        for (int mt = 0; mt < 2; ++mt) {
            ah[mt] = load_frag(sAhi, wr0 + mt * 16, lane);
            al[mt] = load_frag(sAlo, wr0 + mt * 16, lane);
        }
#pragma unroll
        for (int nt = 0; nt < 4; ++nt) {
            v16bf bh = load_frag(sBhi, wc0 + nt * 16, lane);
            v16bf bl = load_frag(sBlo, wc0 + nt * 16, lane);
#pragma unroll
            for (int mt = 0; mt < 2; ++mt) {
                acc[mt][nt] = wmma_bf16(ah[mt], bh, acc[mt][nt]);
                acc[mt][nt] = wmma_bf16(ah[mt], bl, acc[mt][nt]);
                acc[mt][nt] = wmma_bf16(al[mt], bh, acc[mt][nt]);
            }
        }
    }

    // --- epilogue: exp(s/tau), then col sums / row sums / diagonal ---
    // C layout: VGPR r, lanes 0-15 -> (M=r, N=lane); lanes 16-31 -> (M=8+r, N=lane-16)
#pragma unroll
    for (int nt = 0; nt < 4; ++nt) {
        float colacc = 0.f;
#pragma unroll
        for (int mt = 0; mt < 2; ++mt) {
            float e[8];
#pragma unroll
            for (int i = 0; i < 8; ++i) e[i] = __expf(acc[mt][nt][i] * INV_TAU);

            // column sum: reduce 8 VGPRs in-lane, then merge the two halves
            float c8 = ((e[0] + e[1]) + (e[2] + e[3])) +
                       ((e[4] + e[5]) + (e[6] + e[7]));
            c8 += __shfl_xor(c8, 16, 32);
            colacc += c8;

            const int rb = rowbase + wr0 + mt * 16;
            const int cb = colbase + wc0 + nt * 16;

            if (want_row) {  // needed only for the a.b^T pass (transpose dir)
#pragma unroll
                for (int r = 0; r < 8; ++r) {
                    float rv = e[r];
                    rv += __shfl_xor(rv, 1, 32);
                    rv += __shfl_xor(rv, 2, 32);
                    rv += __shfl_xor(rv, 4, 32);
                    rv += __shfl_xor(rv, 8, 32);
                    if (lane == 0)       atomicAdd(&rowsum[rb + r], rv);
                    else if (lane == 16) atomicAdd(&rowsum[rb + 8 + r], rv);
                }
            }
            if (rb == cb) {  // tile contains diagonal elements
#pragma unroll
                for (int r = 0; r < 8; ++r) {
                    if (lane == r)      diag[rb + r]     = e[r];
                    if (lane == 24 + r) diag[rb + 8 + r] = e[r];
                }
            }
        }
        if (lane < 16)
            atomicAdd(&colsum[colbase + wc0 + nt * 16 + lane], colacc);
    }
}

// ===========================================================================
// Kernel 3: assemble per-row losses and mean-reduce to a scalar
// ===========================================================================
__global__ void __launch_bounds__(256)
finalize_loss(const float* __restrict__ rs_aa, const float* __restrict__ rs_ab,
              const float* __restrict__ cs_ab, const float* __restrict__ rs_bb,
              const float* __restrict__ d_aa,  const float* __restrict__ d_ab,
              const float* __restrict__ d_bb,  float* __restrict__ out) {
    float accv = 0.f;
    for (int i = threadIdx.x; i < N_ROWS; i += 256) {
        float den1 = rs_aa[i] + rs_ab[i] - d_aa[i];
        float den2 = rs_bb[i] + cs_ab[i] - d_bb[i];
        // 0.5*(-log(d/den1) - log(d/den2))
        accv += 0.5f * (__logf(den1) + __logf(den2)) - __logf(d_ab[i]);
    }
#pragma unroll
    for (int m = 16; m >= 1; m >>= 1) accv += __shfl_xor(accv, m, 32);
    __shared__ float red[8];
    const int lane = threadIdx.x & 31, wid = threadIdx.x >> 5;
    if (lane == 0) red[wid] = accv;
    __syncthreads();
    if (threadIdx.x == 0) {
        float tot = 0.f;
#pragma unroll
        for (int i = 0; i < 8; ++i) tot += red[i];
        out[0] = tot / (float)N_ROWS;
    }
}

// ===========================================================================
extern "C" void kernel_launch(void* const* d_in, const int* in_sizes, int n_in,
                              void* d_out, int out_size, void* d_ws, size_t ws_size,
                              hipStream_t stream) {
    const float* z1 = (const float*)d_in[0];
    const float* z2 = (const float*)d_in[1];
    float* out = (float*)d_out;

    char* ws = (char*)d_ws;
    const size_t MAT = (size_t)N_ROWS * D_DIM * sizeof(u16);   // 8 MB each
    u16* ahi = (u16*)(ws);
    u16* alo = (u16*)(ws + 1 * MAT);
    u16* bhi = (u16*)(ws + 2 * MAT);
    u16* blo = (u16*)(ws + 3 * MAT);
    float* sums  = (float*)(ws + 4 * MAT);
    float* rs_aa = sums + 0 * N_ROWS;
    float* rs_ab = sums + 1 * N_ROWS;
    float* cs_ab = sums + 2 * N_ROWS;
    float* rs_bb = sums + 3 * N_ROWS;
    float* d_aa  = sums + 4 * N_ROWS;
    float* d_ab  = sums + 5 * N_ROWS;
    float* d_bb  = sums + 6 * N_ROWS;

    // atomically-accumulated vectors must start at zero every call
    (void)hipMemsetAsync(sums, 0, (size_t)4 * N_ROWS * sizeof(float), stream);

    normalize_split<<<dim3(N_ROWS, 2), 256, 0, stream>>>(z1, z2, ahi, alo, bhi, blo);

    dim3 g(N_ROWS / 128, N_ROWS / 128);  // 64 x 64 blocks
    // a.a^T : symmetric -> col sums == row sums
    sim_gemm<<<g, 256, 0, stream>>>(ahi, alo, ahi, alo, rs_aa, nullptr, d_aa, 0);
    // a.b^T : need row sums (for l1) and col sums (== row sums of b.a^T, for l2)
    sim_gemm<<<g, 256, 0, stream>>>(ahi, alo, bhi, blo, cs_ab, rs_ab, d_ab, 1);
    // b.b^T : symmetric
    sim_gemm<<<g, 256, 0, stream>>>(bhi, blo, bhi, blo, rs_bb, nullptr, d_bb, 0);

    finalize_loss<<<1, 256, 0, stream>>>(rs_aa, rs_ab, cs_ab, rs_bb,
                                         d_aa, d_ab, d_bb, out);
}
